// DARE_64622077935667
// MI455X (gfx1250) — matrix operand order repaired and checked by reading, not compiled, not run on hardware
//
#include <hip/hip_runtime.h>
#include <hip/hip_bf16.h>

#define Bsz 2048
#define Lsz 1024
#define Dsz 64
#define Ksz 80
#define Hn  4
#define FFNsz 256
#define NL  2

// per-layer bf16 weight pack offsets (elements)
#define WPACK_L   49152
#define WOFF_WI   0
#define WOFF_WO   12288
#define WOFF_W1   16384
#define WOFF_W2   32768
#define WTOT      (NL * WPACK_L)   // 98304 bf16 elements

typedef __attribute__((ext_vector_type(16))) __bf16 v16bf;
typedef __attribute__((ext_vector_type(8)))  float  v8f;

#define WMMA_BF16(a, b, c) \
  __builtin_amdgcn_wmma_f32_16x16x32_bf16(false, (a), false, (b), (short)0, (c), false, false)

union BF16x16 { uint4 q[2]; v16bf v; };

__device__ __forceinline__ float wave_max32(float v) {
#pragma unroll
  for (int m = 16; m > 0; m >>= 1) v = fmaxf(v, __shfl_xor(v, m, 32));
  return v;
}
__device__ __forceinline__ float wave_sum32(float v) {
#pragma unroll
  for (int m = 16; m > 0; m >>= 1) v += __shfl_xor(v, m, 32);
  return v;
}

// ---- WMMA bf16 fragment loaders: each = two contiguous 16B vector loads ----
// A 16x32 (row m per lane): lane<16 holds K={0..7,16..23}; lane>=16 K={8..15,24..31}
__device__ __forceinline__ v16bf ldA_bf(const __bf16* A, int lda, int m16, int kk) {
  int lane = threadIdx.x & 31, half = lane >> 4, r = lane & 15;
  const __bf16* p = A + (m16 + r) * lda + kk + half * 8;
  BF16x16 u;
  u.q[0] = *(const uint4*)(p);        // K = base+0..7
  u.q[1] = *(const uint4*)(p + 16);   // K = base+16..23
  return u.v;
}
// B 32x16 for C[m][n]=sum_k A[m][k]*W[n][k], W row-major [n][k]:
// lane n holds column n; lane<16 K={0..7,16..23}; lane>=16 K={8..15,24..31}
__device__ __forceinline__ v16bf ldB_bf(const __bf16* W, int ldw, int n16, int kk) {
  int lane = threadIdx.x & 31, half = lane >> 4, n = n16 + (lane & 15);
  const __bf16* p = W + n * ldw + kk + half * 8;
  BF16x16 u;
  u.q[0] = *(const uint4*)(p);
  u.q[1] = *(const uint4*)(p + 16);
  return u.v;
}

// ========== Kernel 0: pack all transformer weights to bf16 ==========
__global__ __launch_bounds__(256) void prep_w_kernel(
    const float* __restrict__ in_w, const float* __restrict__ out_w,
    const float* __restrict__ w1, const float* __restrict__ w2,
    __bf16* __restrict__ dst) {
  int i = blockIdx.x * 256 + threadIdx.x;  // [0, WTOT)
  int l = i / WPACK_L, r = i % WPACK_L;
  float v;
  if (r < WOFF_WO)        v = in_w[(size_t)l * 12288 + r];
  else if (r < WOFF_W1)   v = out_w[(size_t)l * 4096 + (r - WOFF_WO)];
  else if (r < WOFF_W2)   v = w1[(size_t)l * 16384 + (r - WOFF_W1)];
  else                    v = w2[(size_t)l * 16384 + (r - WOFF_W2)];
  dst[i] = (__bf16)v;
}

// ========== Kernel 1: fused gather + decayed dot score ==========
__global__ __launch_bounds__(256) void scores_kernel(
    const int* __restrict__ seq_ids, const float* __restrict__ qv,
    const float* __restrict__ emb_att, float* __restrict__ scores) {
  int b = blockIdx.y;
  int l = blockIdx.x * 256 + threadIdx.x;
  __shared__ float q[Dsz];
  if (threadIdx.x < Dsz) q[threadIdx.x] = qv[b * Dsz + threadIdx.x];
  __syncthreads();
  int id = seq_ids[(size_t)b * Lsz + l];
  const float4* row = (const float4*)(emb_att + (size_t)id * Dsz);
  float acc = 0.0f;
#pragma unroll
  for (int i = 0; i < 16; ++i) {
    float4 r = row[i];
    acc += r.x * q[4 * i] + r.y * q[4 * i + 1] + r.z * q[4 * i + 2] + r.w * q[4 * i + 3];
  }
  float decay = __expf(-(float)(Lsz - 1 - l) * (1.0f / 256.0f));
  float sc = acc + __logf(decay + 1e-8f);
  if (id == 0) sc = -1e9f;  // PAD mask
  scores[(size_t)b * Lsz + l] = sc;
}

// ========== Kernel 2: bitonic top-k -> vals + winning token ids ==========
__global__ __launch_bounds__(256) void topk_kernel(
    const float* __restrict__ scores, const int* __restrict__ seq_ids,
    float* __restrict__ vals, int* __restrict__ tok) {
  int b = blockIdx.x;
  __shared__ unsigned long long key[Lsz];
  const float* s = scores + (size_t)b * Lsz;
  for (int i = threadIdx.x; i < Lsz; i += 256) {
    unsigned u = __float_as_uint(s[i]);
    u = (u & 0x80000000u) ? ~u : (u | 0x80000000u);  // order-preserving map
    key[i] = ((unsigned long long)u << 32) | (unsigned)i;
  }
  __syncthreads();
  for (int k = 2; k <= Lsz; k <<= 1) {
    for (int j = k >> 1; j > 0; j >>= 1) {
      for (int i = threadIdx.x; i < Lsz; i += 256) {
        int ixj = i ^ j;
        if (ixj > i) {
          unsigned long long a = key[i], c = key[ixj];
          bool up = ((i & k) == 0);
          bool sw = up ? (a < c) : (a > c);  // descending sort
          if (sw) { key[i] = c; key[ixj] = a; }
        }
      }
      __syncthreads();
    }
  }
  if (threadIdx.x < Ksz) {
    unsigned long long kk = key[threadIdx.x];
    unsigned enc = (unsigned)(kk >> 32);
    unsigned u = (enc & 0x80000000u) ? (enc & 0x7fffffffu) : ~enc;
    vals[(size_t)b * Ksz + threadIdx.x] = __uint_as_float(u);
    int l = (int)(kk & 0xffffffffu);
    tok[(size_t)b * Ksz + threadIdx.x] = seq_ids[(size_t)b * Lsz + l];
  }
}

// ========== Kernel 3: 2-layer transformer + gating, one block per batch elem ==========
__global__ __launch_bounds__(256) void xformer_kernel(
    const int* __restrict__ tok, const float* __restrict__ vals,
    const float* __restrict__ emb_rep, const __bf16* __restrict__ wbf,
    const float* __restrict__ in_b, const float* __restrict__ out_b,
    const float* __restrict__ n1w, const float* __restrict__ b1,
    const float* __restrict__ b2, const float* __restrict__ n2w,
    const float* __restrict__ rel,
    const float* __restrict__ aux_w, const float* __restrict__ aux_b,
    float* __restrict__ out_u, float* __restrict__ out_aux) {
  int b = blockIdx.x;
  int tid = threadIdx.x, wave = tid >> 5, lane = tid & 31;
  int half = lane >> 4, c = lane & 15;

  __shared__ __align__(16) float  xf[Ksz * Dsz];   // f32 residual stream
  __shared__ __align__(16) __bf16 xb[Ksz * Dsz];   // bf16 mirror (WMMA A feed)
  __shared__ __align__(16) __bf16 obf[Ksz * Dsz];  // attention output
  // persistent per-head buffers; pad regions zeroed ONCE (GEMM epilogues never touch them)
  __shared__ __align__(16) __bf16 qp[Ksz * 32];    // q, K padded 16->32
  __shared__ __align__(16) __bf16 kp[Ksz * 32];    // k, K padded 16->32
  __shared__ __align__(16) __bf16 vT[16 * 96];     // v^T, j padded 80->96
  __shared__ float m161[2 * Ksz + 1];
  __shared__ float wk[Ksz];
  __shared__ float ubuf[Dsz];
  __shared__ __align__(16) union ArenaU {
    struct AttS {
      float  lg[Ksz * Ksz];  // logits
      __bf16 ap[Ksz * 96];   // probs, j padded 80->96 (pads rewritten every head)
    } att;
    struct FfnS { __bf16 f1b[Ksz * FFNsz]; } ffn;
  } ar;

  // zero qp/kp/vT once (contiguous-ish; just blanket all three buffers)
  {
    int* z = (int*)qp;
    for (int i = tid; i < (Ksz * 32) / 2, i < 1280; i += 256) z[i] = 0;
    int* z2 = (int*)kp;
    for (int i = tid; i < 1280; i += 256) z2[i] = 0;
    int* z3 = (int*)vT;
    for (int i = tid; i < 768; i += 256) z3[i] = 0;
  }

  // ---- async gather of rep rows straight into LDS (GV mode, per-lane addr) ----
  {
    unsigned base = (unsigned)(uintptr_t)(void*)xf;  // LDS byte offset of xf
#pragma unroll
    for (int i = 0; i < (Ksz * Dsz) / (4 * 256); ++i) {  // 5 iterations of 256 lanes x 16B
      int t = tid + i * 256;                 // vec4 chunk index
      int k = (t * 4) >> 6, d4 = (t * 4) & 63;
      int id = tok[(size_t)b * Ksz + k];
      const float* gp = emb_rep + (size_t)id * Dsz + d4;
      unsigned laddr = base + (unsigned)t * 16u;
      asm volatile("global_load_async_to_lds_b128 %0, %1, off"
                   :: "v"(laddr), "v"(gp)
                   : "memory");
    }
    asm volatile("s_wait_asynccnt 0x0" ::: "memory");
  }
  __syncthreads();
  for (int t = tid; t < Ksz * Dsz; t += 256) xb[t] = (__bf16)xf[t];
  __syncthreads();

  for (int l = 0; l < NL; ++l) {
    const __bf16* Wi = wbf + (size_t)l * WPACK_L + WOFF_WI;  // [192][64]
    const __bf16* Wo = wbf + (size_t)l * WPACK_L + WOFF_WO;  // [64][64]
    const __bf16* W1 = wbf + (size_t)l * WPACK_L + WOFF_W1;  // [256][64]
    const __bf16* W2 = wbf + (size_t)l * WPACK_L + WOFF_W2;  // [64][256]
    const float* bi = in_b  + (size_t)l * 3 * Dsz;
    const float* bo = out_b + (size_t)l * Dsz;
    const float* B1 = b1 + (size_t)l * FFNsz;
    const float* B2 = b2 + (size_t)l * Dsz;

    for (int p = tid; p < 2 * Ksz + 1; p += 256) {
      const float* rp = rel + ((size_t)l * (2 * Ksz + 1) + p) * Hn;
      m161[p] = 0.25f * (rp[0] + rp[1] + rp[2] + rp[3]);
    }
    __syncthreads();

    // ---- attention, one head at a time ----
    for (int h = 0; h < Hn; ++h) {
      // QKV slice GEMM (per head): 15 tiles, K=64
      for (int t0 = wave; t0 < 15; t0 += 8) {
        int t = __builtin_amdgcn_readfirstlane(t0);
        int which = t / 5, mt = t % 5, m16 = mt * 16;
        int n0 = which * 64 + h * 16;
        v8f acc = {};
#pragma unroll
        for (int kk = 0; kk < 64; kk += 32)
          acc = WMMA_BF16(ldA_bf(xb, Dsz, m16, kk), ldB_bf(Wi, Dsz, n0, kk), acc);
        float bb = bi[n0 + c];
        if (which == 0) {
#pragma unroll
          for (int r = 0; r < 8; ++r)
            qp[(m16 + r + 8 * half) * 32 + c] = (__bf16)(acc[r] + bb);
        } else if (which == 1) {
#pragma unroll
          for (int r = 0; r < 8; ++r)
            kp[(m16 + r + 8 * half) * 32 + c] = (__bf16)(acc[r] + bb);
        } else {
#pragma unroll
          for (int r = 0; r < 8; ++r)
            vT[c * 96 + (m16 + r + 8 * half)] = (__bf16)(acc[r] + bb);
        }
      }
      __syncthreads();

      // logits = q @ k^T * scale + rel-bias (K padded to 32)
      for (int t0 = wave; t0 < 25; t0 += 8) {
        int t = __builtin_amdgcn_readfirstlane(t0);
        int mt = t / 5, nt = t % 5, m16 = mt * 16;
        v8f acc = {};
        acc = WMMA_BF16(ldA_bf(qp, 32, m16, 0), ldB_bf(kp, 32, nt * 16, 0), acc);
#pragma unroll
        for (int r = 0; r < 8; ++r) {
          int mq = m16 + r + 8 * half;
          int j = nt * 16 + c;
          ar.att.lg[mq * Ksz + j] = acc[r] * 0.25f + m161[j - mq + Ksz];
        }
      }
      __syncthreads();

      // row softmax -> bf16 probs (cols 80..95 zeroed by lanes 16..31 every head)
      for (int r = wave; r < Ksz; r += 8) {
        const float* row = ar.att.lg + r * Ksz;
        float v0 = row[lane], v1 = row[lane + 32];
        float v2 = (lane + 64 < Ksz) ? row[lane + 64] : -3.4e38f;
        float mx = wave_max32(fmaxf(fmaxf(v0, v1), v2));
        float e0 = __expf(v0 - mx), e1 = __expf(v1 - mx);
        float e2 = (lane + 64 < Ksz) ? __expf(v2 - mx) : 0.0f;
        float inv = 1.0f / wave_sum32(e0 + e1 + e2);
        __bf16* arow = ar.att.ap + r * 96;
        arow[lane]      = (__bf16)(e0 * inv);
        arow[lane + 32] = (__bf16)(e1 * inv);
        arow[lane + 64] = (__bf16)(e2 * inv);
      }
      __syncthreads();

      // o = a @ v (K padded 80->96)
      for (int t0 = wave; t0 < 5; t0 += 8) {
        int m16 = __builtin_amdgcn_readfirstlane(t0) * 16;
        v8f acc = {};
#pragma unroll
        for (int kk = 0; kk < 96; kk += 32)
          acc = WMMA_BF16(ldA_bf(ar.att.ap, 96, m16, kk), ldB_bf(vT, 96, 0, kk), acc);
#pragma unroll
        for (int r = 0; r < 8; ++r)
          obf[(m16 + r + 8 * half) * Dsz + h * 16 + c] = (__bf16)acc[r];
      }
      __syncthreads();
    }

    // out-proj + residual into xf
    for (int t0 = wave; t0 < 20; t0 += 8) {
      int t = __builtin_amdgcn_readfirstlane(t0);
      int m16 = (t >> 2) * 16, n0 = (t & 3) * 16;
      v8f acc = {};
#pragma unroll
      for (int kk = 0; kk < 64; kk += 32)
        acc = WMMA_BF16(ldA_bf(obf, Dsz, m16, kk), ldB_bf(Wo, Dsz, n0, kk), acc);
      float bb = bo[n0 + c];
#pragma unroll
      for (int r = 0; r < 8; ++r)
        xf[(m16 + r + 8 * half) * Dsz + n0 + c] += acc[r] + bb;
    }
    __syncthreads();

    // rmsnorm1 (updates f32 master + bf16 mirror)
    {
      const float* wn = n1w + (size_t)l * Dsz;
      for (int r = wave; r < Ksz; r += 8) {
        float a0 = xf[r * Dsz + lane], a1 = xf[r * Dsz + lane + 32];
        float ss = wave_sum32(a0 * a0 + a1 * a1);
        float rr = rsqrtf(ss * (1.0f / (float)Dsz) + 1e-6f);
        float y0 = wn[lane] * a0 * rr, y1 = wn[lane + 32] * a1 * rr;
        xf[r * Dsz + lane] = y0;       xb[r * Dsz + lane] = (__bf16)y0;
        xf[r * Dsz + lane + 32] = y1;  xb[r * Dsz + lane + 32] = (__bf16)y1;
      }
    }
    __syncthreads();

    // FFN1: gelu(x @ w1^T + b1) -> bf16
    for (int t0 = wave; t0 < 80; t0 += 8) {
      int t = __builtin_amdgcn_readfirstlane(t0);
      int m16 = (t % 5) * 16, n0 = (t / 5) * 16;
      v8f acc = {};
#pragma unroll
      for (int kk = 0; kk < 64; kk += 32)
        acc = WMMA_BF16(ldA_bf(xb, Dsz, m16, kk), ldB_bf(W1, Dsz, n0, kk), acc);
      float bb = B1[n0 + c];
#pragma unroll
      for (int r = 0; r < 8; ++r) {
        float z = acc[r] + bb;
        float g = 0.5f * z * (1.0f + erff(z * 0.70710678f));  // exact gelu
        ar.ffn.f1b[(m16 + r + 8 * half) * FFNsz + n0 + c] = (__bf16)g;
      }
    }
    __syncthreads();

    // FFN2 + residual into xf
    for (int t0 = wave; t0 < 20; t0 += 8) {
      int t = __builtin_amdgcn_readfirstlane(t0);
      int m16 = (t >> 2) * 16, n0 = (t & 3) * 16;
      v8f acc = {};
#pragma unroll
      for (int kk = 0; kk < FFNsz; kk += 32)
        acc = WMMA_BF16(ldA_bf(ar.ffn.f1b, FFNsz, m16, kk), ldB_bf(W2, FFNsz, n0, kk), acc);
      float bb = B2[n0 + c];
#pragma unroll
      for (int r = 0; r < 8; ++r)
        xf[(m16 + r + 8 * half) * Dsz + n0 + c] += acc[r] + bb;
    }
    __syncthreads();

    // rmsnorm2
    {
      const float* wn = n2w + (size_t)l * Dsz;
      for (int r = wave; r < Ksz; r += 8) {
        float a0 = xf[r * Dsz + lane], a1 = xf[r * Dsz + lane + 32];
        float ss = wave_sum32(a0 * a0 + a1 * a1);
        float rr = rsqrtf(ss * (1.0f / (float)Dsz) + 1e-6f);
        float y0 = wn[lane] * a0 * rr, y1 = wn[lane + 32] * a1 * rr;
        xf[r * Dsz + lane] = y0;       xb[r * Dsz + lane] = (__bf16)y0;
        xf[r * Dsz + lane + 32] = y1;  xb[r * Dsz + lane + 32] = (__bf16)y1;
      }
    }
    __syncthreads();
  }

  // ---- gating: w = softmax(vals), u = sum_k w_k * x_k ----
  if (wave == 0) {
    const float* vrow = vals + (size_t)b * Ksz;
    float v0 = vrow[lane], v1 = vrow[lane + 32];
    float v2 = (lane + 64 < Ksz) ? vrow[lane + 64] : -3.4e38f;
    float mx = wave_max32(fmaxf(fmaxf(v0, v1), v2));
    float e0 = __expf(v0 - mx), e1 = __expf(v1 - mx);
    float e2 = (lane + 64 < Ksz) ? __expf(v2 - mx) : 0.0f;
    float inv = 1.0f / wave_sum32(e0 + e1 + e2);
    wk[lane] = e0 * inv;
    wk[lane + 32] = e1 * inv;
    if (lane + 64 < Ksz) wk[lane + 64] = e2 * inv;
  }
  __syncthreads();
  if (tid < Dsz) {
    float acc = 0.0f;
    for (int k = 0; k < Ksz; ++k) acc += wk[k] * xf[k * Dsz + tid];
    ubuf[tid] = acc;
    out_u[(size_t)b * Dsz + tid] = acc;
  }
  __syncthreads();
  if (wave == 0) {
    float p = ubuf[lane] * aux_w[lane] + ubuf[lane + 32] * aux_w[lane + 32];
    p = wave_sum32(p);
    if (lane == 0) out_aux[b] = p + aux_b[0];
  }
}

extern "C" void kernel_launch(void* const* d_in, const int* in_sizes, int n_in,
                              void* d_out, int out_size, void* d_ws, size_t ws_size,
                              hipStream_t stream) {
  (void)in_sizes; (void)n_in; (void)out_size; (void)ws_size;
  const int*   seq_ids = (const int*)  d_in[0];
  const float* qv      = (const float*)d_in[1];
  const float* emb_att = (const float*)d_in[2];
  const float* emb_rep = (const float*)d_in[3];
  const float* in_w    = (const float*)d_in[4];
  const float* in_b    = (const float*)d_in[5];
  const float* out_w   = (const float*)d_in[6];
  const float* out_b   = (const float*)d_in[7];
  const float* n1w     = (const float*)d_in[8];
  const float* w1      = (const float*)d_in[9];
  const float* b1      = (const float*)d_in[10];
  const float* w2      = (const float*)d_in[11];
  const float* b2      = (const float*)d_in[12];
  const float* n2w     = (const float*)d_in[13];
  const float* rel     = (const float*)d_in[14];
  const float* aux_w   = (const float*)d_in[15];
  const float* aux_b   = (const float*)d_in[16];

  float*  scores = (float*)d_ws;                     // B*L f32
  float*  vals   = scores + (size_t)Bsz * Lsz;       // B*K f32
  int*    tok    = (int*)(vals + (size_t)Bsz * Ksz); // B*K i32
  __bf16* wbf    = (__bf16*)(tok + (size_t)Bsz * Ksz); // WTOT bf16 (16B-aligned)

  float* out_u   = (float*)d_out;                    // B*D
  float* out_aux = out_u + (size_t)Bsz * Dsz;        // B

  prep_w_kernel<<<WTOT / 256, 256, 0, stream>>>(in_w, out_w, w1, w2, wbf);
  scores_kernel<<<dim3(Lsz / 256, Bsz), 256, 0, stream>>>(seq_ids, qv, emb_att, scores);
  topk_kernel<<<Bsz, 256, 0, stream>>>(scores, seq_ids, vals, tok);
  xformer_kernel<<<Bsz, 256, 0, stream>>>(tok, vals, emb_rep, wbf, in_b, out_b, n1w,
                                          b1, b2, n2w, rel, aux_w, aux_b,
                                          out_u, out_aux);
}
